// MoELayer_27513560498336
// MI455X (gfx1250) — matrix operand order repaired and checked by reading, not compiled, not run on hardware
//
#include <hip/hip_runtime.h>
#include <math.h>

// ---------------------------------------------------------------------------
// MoE FFN for MI455X (gfx1250): sparse top-2 dispatch + bf16 WMMA GEMMs.
// A tiles: global_load_async_to_lds_b128 (ASYNCcnt). B tiles: fp32->bf16
// k-paired packs (v_cvt_pk_bf16_f32) -> ds_store_b32 transposed.
// ---------------------------------------------------------------------------

typedef __bf16 bf16_t;
typedef bf16_t v16bf __attribute__((ext_vector_type(16)));
typedef bf16_t v8bf  __attribute__((ext_vector_type(8)));
typedef bf16_t v2bf  __attribute__((ext_vector_type(2)));
typedef float  v8f   __attribute__((ext_vector_type(8)));

// Problem constants (match reference)
constexpr int N_TOK = 8192;        // B*T
constexpr int C_DIM = 1024;
constexpr int DFF   = 4096;
constexpr int NEXP  = 8;

// GEMM tiling
constexpr int BM  = 128;
constexpr int BN  = 128;
constexpr int BK  = 32;
constexpr int LDT = 40;            // padded LDS stride (bf16 elems): 16B-aligned frags, conflict-free

constexpr int MPAD   = 2 * N_TOK + NEXP * BM;   // 17408: worst-case padded rows
constexpr int MTILES = MPAD / BM;               // 136

// Workspace layout (bytes)
constexpr size_t OFF_COUNTS = 0;
constexpr size_t OFF_POFF   = 256;
constexpr size_t OFF_TOK    = 512;                                   // E*N ints   = 256 KB
constexpr size_t OFF_WGT    = OFF_TOK + (size_t)NEXP * N_TOK * 4;    // E*N floats = 256 KB
constexpr size_t OFF_RW     = OFF_WGT + (size_t)NEXP * N_TOK * 4;    // MPAD floats
constexpr size_t OFF_XG     = (OFF_RW + (size_t)MPAD * 4 + 255) & ~(size_t)255;  // MPAD*C bf16
constexpr size_t OFF_H      = (OFF_XG + (size_t)MPAD * C_DIM * 2 + 255) & ~(size_t)255; // MPAD*DFF bf16

// ---------------------------------------------------------------------------
// 0) Zero output (harness poisons it) + expert counters
// ---------------------------------------------------------------------------
__global__ __launch_bounds__(256) void moe_zero(float* __restrict__ out, int* __restrict__ counts) {
    const size_t nt = (size_t)gridDim.x * blockDim.x;
    size_t i = (size_t)blockIdx.x * blockDim.x + threadIdx.x;
    float4 z = make_float4(0.f, 0.f, 0.f, 0.f);
    float4* o4 = (float4*)out;
    const size_t n4 = (size_t)N_TOK * C_DIM / 4;
    for (size_t k = i; k < n4; k += nt) o4[k] = z;
    if (i < NEXP) counts[i] = 0;
}

// ---------------------------------------------------------------------------
// 1) Router: one wave32 per token. logits -> softmax -> top-2 -> slot append.
// ---------------------------------------------------------------------------
__global__ __launch_bounds__(256) void moe_router(
    const float* __restrict__ x, const float* __restrict__ rw,
    const float* __restrict__ rb, int* __restrict__ counts,
    int* __restrict__ tok, float* __restrict__ wgt)
{
    const int wid  = threadIdx.x >> 5;
    const int lane = threadIdx.x & 31;
    const int t = blockIdx.x * 8 + wid;
    if (t >= N_TOK) return;

    const float* xr = x + (size_t)t * C_DIM;
    float acc[NEXP];
#pragma unroll
    for (int e = 0; e < NEXP; ++e) acc[e] = 0.f;

    for (int c = lane; c < C_DIM; c += 32) {
        const float xv = xr[c];
        const float4* rwp = (const float4*)(rw + (size_t)c * NEXP);
        const float4 r0 = rwp[0], r1 = rwp[1];
        acc[0] += xv * r0.x; acc[1] += xv * r0.y; acc[2] += xv * r0.z; acc[3] += xv * r0.w;
        acc[4] += xv * r1.x; acc[5] += xv * r1.y; acc[6] += xv * r1.z; acc[7] += xv * r1.w;
    }
    // wave32 butterfly reduction
#pragma unroll
    for (int off = 16; off >= 1; off >>= 1)
#pragma unroll
        for (int e = 0; e < NEXP; ++e) acc[e] += __shfl_xor(acc[e], off, 32);

    // softmax over 8 logits (all lanes redundantly compute the same values)
    float m = -INFINITY;
#pragma unroll
    for (int e = 0; e < NEXP; ++e) { acc[e] += rb[e]; m = fmaxf(m, acc[e]); }
    float s = 0.f;
#pragma unroll
    for (int e = 0; e < NEXP; ++e) { acc[e] = expf(acc[e] - m); s += acc[e]; }
    const float inv = 1.f / s;

    // top-2 (ties -> lowest index, matching lax.top_k)
    int e0 = 0; float p0 = acc[0];
#pragma unroll
    for (int e = 1; e < NEXP; ++e) if (acc[e] > p0) { p0 = acc[e]; e0 = e; }
    int e1 = -1; float p1 = -INFINITY;
#pragma unroll
    for (int e = 0; e < NEXP; ++e) if (e != e0 && acc[e] > p1) { p1 = acc[e]; e1 = e; }

    if (lane == 0) {
        int s0 = atomicAdd(&counts[e0], 1);
        tok[e0 * N_TOK + s0] = t; wgt[e0 * N_TOK + s0] = p0 * inv;
        int s1 = atomicAdd(&counts[e1], 1);
        tok[e1 * N_TOK + s1] = t; wgt[e1 * N_TOK + s1] = p1 * inv;
    }
}

// ---------------------------------------------------------------------------
// 2) Prefix: padded expert row offsets (each region a multiple of BM)
// ---------------------------------------------------------------------------
__global__ void moe_prefix(const int* __restrict__ counts, int* __restrict__ poff) {
    if (threadIdx.x == 0 && blockIdx.x == 0) {
        int a = 0;
        for (int e = 0; e < NEXP; ++e) {
            poff[e] = a;
            a += (counts[e] + BM - 1) & ~(BM - 1);
        }
        poff[NEXP] = a;
    }
}

// ---------------------------------------------------------------------------
// 3) Gather: pack selected token rows (fp32 -> bf16) + per-row weight
// ---------------------------------------------------------------------------
__global__ __launch_bounds__(256) void moe_gather(
    const float* __restrict__ x, const int* __restrict__ tok,
    const float* __restrict__ wgt, const int* __restrict__ counts,
    const int* __restrict__ poff, bf16_t* __restrict__ Xg,
    float* __restrict__ rowWeight)
{
    const int e    = blockIdx.x >> 13;       // blockIdx.x = e*N_TOK + slot
    const int slot = blockIdx.x & (N_TOK - 1);
    if (slot >= counts[e]) return;
    const int row = poff[e] + slot;
    const int t   = tok[e * N_TOK + slot];
    const float4 v = ((const float4*)(x + (size_t)t * C_DIM))[threadIdx.x];
    bf16_t o[4] = { (bf16_t)v.x, (bf16_t)v.y, (bf16_t)v.z, (bf16_t)v.w };
    *(uint2*)&Xg[(size_t)row * C_DIM + threadIdx.x * 4] = *(uint2*)o;
    if (threadIdx.x == 0) rowWeight[row] = wgt[e * N_TOK + slot];
}

// ---------------------------------------------------------------------------
// Tile -> expert lookup (uniform per block)
// ---------------------------------------------------------------------------
__device__ __forceinline__ void tile_expert(const int* counts, const int* poff,
                                            int m0, int* sInfo) {
    int e = -1, base = 0;
    if (m0 < poff[NEXP]) {
        for (int i = 0; i < NEXP; ++i)
            if (m0 >= poff[i] && m0 < poff[i + 1]) { e = i; base = poff[i]; break; }
    }
    sInfo[0] = e;
    sInfo[1] = (e >= 0) ? counts[e] : 0;
    sInfo[2] = base;
}

// Build a v16bf WMMA fragment from two aligned 16B LDS reads
__device__ __forceinline__ v16bf frag16(const bf16_t* p, int off2) {
    v8bf lo = *(const v8bf*)p;
    v8bf hi = *(const v8bf*)(p + off2);
    return __builtin_shufflevector(lo, hi, 0,1,2,3,4,5,6,7,8,9,10,11,12,13,14,15);
}

// ---------------------------------------------------------------------------
// Shared GEMM main loop.
//   A: bf16, row-major, leading dim LDA_, copied global->LDS via async DMA.
//   B: fp32, row-major (k-major), leading dim LDB_, converted to bf16 and
//      stored transposed (n-major) in LDS as packed k-pairs.
// ---------------------------------------------------------------------------
template<int LDA_, int LDB_, int KSTEPS_>
__device__ __forceinline__ void gemm_mainloop(
    const bf16_t* __restrict__ Ag, const float* __restrict__ Be,
    int m0, int n0, bf16_t* As, bf16_t* Bt, v8f (&acc)[4][2], int tid)
{
    const int wid = tid >> 5, lane = tid & 31;
    const int wm = wid >> 2, wn = wid & 3;     // 2x4 wave grid
    const int lm = lane & 15, lh = lane >> 4;
    const int ar = tid >> 1, ah = (tid & 1) * 16;

    for (int kt = 0; kt < KSTEPS_; ++kt) {
        const int k0 = kt * BK;

        // B tile: per thread 2 work items, each a k/k+1 pair of float4 (4 cols)
        float b0a[2][4], b1a[2][4]; int kp[2], c4[2];
#pragma unroll
        for (int i = 0; i < 2; ++i) {
            const int p = tid + 256 * i;       // 0..511
            kp[i] = (p >> 5) * 2;              // even k within tile: 0..30
            c4[i] = (p & 31) * 4;              // n within tile: 0..124
            const float* bp = Be + (size_t)(k0 + kp[i]) * LDB_ + n0 + c4[i];
            const float4 t0 = *(const float4*)bp;
            const float4 t1 = *(const float4*)(bp + LDB_);
            b0a[i][0] = t0.x; b0a[i][1] = t0.y; b0a[i][2] = t0.z; b0a[i][3] = t0.w;
            b1a[i][0] = t1.x; b1a[i][1] = t1.y; b1a[i][2] = t1.z; b1a[i][3] = t1.w;
        }
        if (kt + 1 < KSTEPS_)                  // global_prefetch_b8 next B K-tile
            __builtin_prefetch(Be + (size_t)(k0 + BK + kp[0]) * LDB_ + n0 + c4[0], 0, 1);

        __syncthreads();                       // previous iteration done reading LDS

        // A tile: async DMA global -> LDS, two b128 (16B) segments per thread.
        {
            const bf16_t* ag = Ag + (size_t)(m0 + ar) * LDA_ + k0 + ah;
            const unsigned lo = (unsigned)(size_t)&As[ar * LDT + ah];
            asm volatile(
                "global_load_async_to_lds_b128 %0, %2, off\n\t"
                "global_load_async_to_lds_b128 %1, %3, off"
                :: "v"(lo), "v"(lo + 16u),
                   "v"((unsigned long long)(size_t)ag),
                   "v"((unsigned long long)(size_t)(ag + 8))
                : "memory");
        }

        // B: pack (k, k+1) bf16 pairs -> single b32 LDS store each
#pragma unroll
        for (int i = 0; i < 2; ++i)
#pragma unroll
            for (int j = 0; j < 4; ++j)
                *(v2bf*)&Bt[(c4[i] + j) * LDT + kp[i]] =
                    (v2bf){ (bf16_t)b0a[i][j], (bf16_t)b1a[i][j] };

        asm volatile("s_wait_asynccnt 0x0" ::: "memory");
        __syncthreads();

        v16bf a[4], b[2];
#pragma unroll
        for (int i = 0; i < 4; ++i)
            a[i] = frag16(&As[(wm * 64 + i * 16 + lm) * LDT + lh * 8], 16);
#pragma unroll
        for (int j = 0; j < 2; ++j)
            b[j] = frag16(&Bt[(wn * 32 + j * 16 + lm) * LDT + lh * 8], 16);
#pragma unroll
        for (int i = 0; i < 4; ++i)
#pragma unroll
            for (int j = 0; j < 2; ++j)
                acc[i][j] = __builtin_amdgcn_wmma_f32_16x16x32_bf16(
                    false, a[i], false, b[j], (short)0, acc[i][j], false, false);
    }
}

// ---------------------------------------------------------------------------
// 4) GEMM1: H = gelu(Xg @ w1[e] + b1[e]), bf16 WMMA, fp32 acc
// ---------------------------------------------------------------------------
__global__ __launch_bounds__(256) void moe_gemm1(
    const bf16_t* __restrict__ Xg, const float* __restrict__ w1,
    const float* __restrict__ b1, bf16_t* __restrict__ H,
    const int* __restrict__ counts, const int* __restrict__ poff)
{
    __shared__ bf16_t As[BM * LDT];
    __shared__ bf16_t Bt[BN * LDT];
    __shared__ int sInfo[3];

    const int tid = threadIdx.x;
    const int m0 = blockIdx.x * BM;
    const int n0 = blockIdx.y * BN;

    if (tid == 0) tile_expert(counts, poff, m0, sInfo);
    __syncthreads();
    const int e = sInfo[0];
    if (e < 0) return;
    const int cnt = sInfo[1], base = sInfo[2];

    v8f acc[4][2];
#pragma unroll
    for (int i = 0; i < 4; ++i)
#pragma unroll
        for (int j = 0; j < 2; ++j) acc[i][j] = (v8f){0.f,0.f,0.f,0.f,0.f,0.f,0.f,0.f};

    gemm_mainloop<C_DIM, DFF, C_DIM / BK>(
        Xg, w1 + (size_t)e * C_DIM * DFF, m0, n0, As, Bt, acc, tid);

    const int wid = tid >> 5, lane = tid & 31;
    const int wm = wid >> 2, wn = wid & 3;
    const int lm = lane & 15, lh = lane >> 4;

    // epilogue: exact GELU, store bf16 H for valid rows only
#pragma unroll
    for (int i = 0; i < 4; ++i) {
        const int gm = m0 + wm * 64 + i * 16 + lh * 8;
#pragma unroll
        for (int j = 0; j < 2; ++j) {
            const int gn = n0 + wn * 32 + j * 16 + lm;
            const float bias = b1[e * DFF + gn];
#pragma unroll
            for (int r = 0; r < 8; ++r) {
                const int row = gm + r;
                if (row - base < cnt) {
                    const float v = acc[i][j][r] + bias;
                    const float g = 0.5f * v * (1.0f + erff(v * 0.70710678118654752f));
                    H[(size_t)row * DFF + gn] = (bf16_t)g;
                }
            }
        }
    }
}

// ---------------------------------------------------------------------------
// 5) GEMM2: out[token] += w_route * (H @ w2[e] + b2[e])
// ---------------------------------------------------------------------------
__global__ __launch_bounds__(256) void moe_gemm2(
    const bf16_t* __restrict__ H, const float* __restrict__ w2,
    const float* __restrict__ b2, const float* __restrict__ rowWeight,
    const int* __restrict__ tok, const int* __restrict__ counts,
    const int* __restrict__ poff, float* __restrict__ out)
{
    __shared__ bf16_t As[BM * LDT];
    __shared__ bf16_t Bt[BN * LDT];
    __shared__ int sInfo[3];

    const int tid = threadIdx.x;
    const int m0 = blockIdx.x * BM;
    const int n0 = blockIdx.y * BN;

    if (tid == 0) tile_expert(counts, poff, m0, sInfo);
    __syncthreads();
    const int e = sInfo[0];
    if (e < 0) return;
    const int cnt = sInfo[1], base = sInfo[2];

    v8f acc[4][2];
#pragma unroll
    for (int i = 0; i < 4; ++i)
#pragma unroll
        for (int j = 0; j < 2; ++j) acc[i][j] = (v8f){0.f,0.f,0.f,0.f,0.f,0.f,0.f,0.f};

    gemm_mainloop<DFF, C_DIM, DFF / BK>(
        H, w2 + (size_t)e * DFF * C_DIM, m0, n0, As, Bt, acc, tid);

    const int wid = tid >> 5, lane = tid & 31;
    const int wm = wid >> 2, wn = wid & 3;
    const int lm = lane & 15, lh = lane >> 4;

    // epilogue: scale by router weight, scatter-accumulate into token rows.
    // Each out element receives exactly 2 adds (fp add is commutative) -> deterministic.
#pragma unroll
    for (int i = 0; i < 4; ++i) {
        const int gm = m0 + wm * 64 + i * 16 + lh * 8;
#pragma unroll
        for (int j = 0; j < 2; ++j) {
            const int gn = n0 + wn * 32 + j * 16 + lm;
            const float bias = b2[e * C_DIM + gn];
#pragma unroll
            for (int r = 0; r < 8; ++r) {
                const int row = gm + r;
                const int lrow = row - base;
                if (lrow < cnt) {
                    const float v = rowWeight[row] * (acc[i][j][r] + bias);
                    const int tkn = tok[e * N_TOK + lrow];
                    atomicAdd(out + (size_t)tkn * C_DIM + gn, v);
                }
            }
        }
    }
}

// ---------------------------------------------------------------------------
// Launch
// ---------------------------------------------------------------------------
extern "C" void kernel_launch(void* const* d_in, const int* in_sizes, int n_in,
                              void* d_out, int out_size, void* d_ws, size_t ws_size,
                              hipStream_t stream) {
    (void)in_sizes; (void)n_in; (void)out_size; (void)ws_size;
    const float* x        = (const float*)d_in[0];
    const float* router_w = (const float*)d_in[1];
    const float* router_b = (const float*)d_in[2];
    const float* w1       = (const float*)d_in[3];
    const float* b1       = (const float*)d_in[4];
    const float* w2       = (const float*)d_in[5];
    const float* b2       = (const float*)d_in[6];
    float* out = (float*)d_out;

    char* ws = (char*)d_ws;
    int*    counts    = (int*)(ws + OFF_COUNTS);
    int*    poff      = (int*)(ws + OFF_POFF);
    int*    tok       = (int*)(ws + OFF_TOK);
    float*  wgt       = (float*)(ws + OFF_WGT);
    float*  rowWeight = (float*)(ws + OFF_RW);
    bf16_t* Xg        = (bf16_t*)(ws + OFF_XG);
    bf16_t* Hbuf      = (bf16_t*)(ws + OFF_H);

    moe_zero  <<<2048, 256, 0, stream>>>(out, counts);
    moe_router<<<N_TOK / 8, 256, 0, stream>>>(x, router_w, router_b, counts, tok, wgt);
    moe_prefix<<<1, 32, 0, stream>>>(counts, poff);
    moe_gather<<<NEXP * N_TOK, 256, 0, stream>>>(x, tok, wgt, counts, poff, Xg, rowWeight);
    moe_gemm1 <<<dim3(MTILES, DFF / BN), 256, 0, stream>>>(Xg, w1, b1, Hbuf, counts, poff);
    moe_gemm2 <<<dim3(MTILES, C_DIM / BN), 256, 0, stream>>>(Hbuf, w2, b2, rowWeight, tok, counts, poff, out);
}
